// _Int8DynActIntxWeightQuantizedLinearNative_1812476199311
// MI455X (gfx1250) — 8-bit matrix operand order repaired and checked by reading, compiled, not run on hardware
//
#include <hip/hip_runtime.h>
#include <stdint.h>

#define M_DIM 8192
#define K_DIM 4096
#define N_DIM 4096
#define GS    128
#define NG    (K_DIM / GS)

#define BROWS   64
#define BPITCH  144

typedef __attribute__((ext_vector_type(8))) int   v8i;
typedef __attribute__((ext_vector_type(8))) float v8f;

static __device__ __forceinline__ float wave_min_f(float v) {
#pragma unroll
  for (int m = 16; m >= 1; m >>= 1) v = fminf(v, __shfl_xor(v, m, 32));
  return v;
}
static __device__ __forceinline__ float wave_max_f(float v) {
#pragma unroll
  for (int m = 16; m >= 1; m >>= 1) v = fmaxf(v, __shfl_xor(v, m, 32));
  return v;
}
static __device__ __forceinline__ int wave_sum_i(int v) {
#pragma unroll
  for (int m = 16; m >= 1; m >>= 1) v += __shfl_xor(v, m, 32);
  return v;
}

__global__ void quant_w_kernel(const float* __restrict__ w,
                               int8_t* __restrict__ qw,
                               float* __restrict__ wscale,
                               float* __restrict__ sqr) {
  const int lane = threadIdx.x & 31;
  const int wid  = threadIdx.x >> 5;
  const int gi   = blockIdx.x * 8 + wid;
  const int n = gi / NG;
  const int g = gi % NG;

  const float4 f = *(const float4*)(w + (size_t)n * K_DIM + (size_t)g * GS + lane * 4);

  float vmin = fminf(fminf(f.x, f.y), fminf(f.z, f.w));
  float vmax = fmaxf(fmaxf(f.x, f.y), fmaxf(f.z, f.w));
  vmin = wave_min_f(vmin);
  vmax = wave_max_f(vmax);

  const float d  = vmax - vmin;
  const float s  = (d > 0.f) ? d * (1.0f / 15.0f) : 1.0f;
  const float rs = 1.0f / s;
  const float zw = -8.0f - rintf(vmin * rs);

  int r[4];
  {
    const float e[4] = {f.x, f.y, f.z, f.w};
#pragma unroll
    for (int i = 0; i < 4; ++i) {
      float q = fminf(fmaxf(rintf(e[i] * rs) + zw, -8.0f), 7.0f);
      int   ri = (int)(q - zw);
      r[i] = ri < -128 ? -128 : (ri > 127 ? 127 : ri);
    }
  }
  const int packed = (r[0] & 0xFF) | ((r[1] & 0xFF) << 8) |
                     ((r[2] & 0xFF) << 16) | ((r[3] & 0xFF) << 24);
  *(int*)(qw + (size_t)n * K_DIM + (size_t)g * GS + lane * 4) = packed;

  const int gsum = wave_sum_i(r[0] + r[1] + r[2] + r[3]);
  if (lane == 0) {
    wscale[gi] = s;
    sqr[gi]    = (float)gsum;
  }
}

__global__ void cw_kernel(const float* __restrict__ wscale,
                          const float* __restrict__ sqr,
                          float* __restrict__ cw) {
  const int n = blockIdx.x * blockDim.x + threadIdx.x;
  float acc = 0.f;
#pragma unroll
  for (int g = 0; g < NG; ++g)
    acc += wscale[(size_t)n * NG + g] * sqr[(size_t)n * NG + g];
  cw[n] = acc;
}

__global__ void quant_x_kernel(const float* __restrict__ x,
                               int8_t* __restrict__ qx,
                               float* __restrict__ xs,
                               float* __restrict__ xo) {
  __shared__ float smin[8];
  __shared__ float smax[8];

  const int m    = blockIdx.x;
  const int t    = threadIdx.x;
  const int lane = t & 31;
  const int wid  = t >> 5;

  const float* row = x + (size_t)m * K_DIM;
  float4 f[4];
#pragma unroll
  for (int i = 0; i < 4; ++i) f[i] = *(const float4*)(row + t * 16 + i * 4);

  float vmin =  3.4e38f, vmax = -3.4e38f;
#pragma unroll
  for (int i = 0; i < 4; ++i) {
    vmin = fminf(vmin, fminf(fminf(f[i].x, f[i].y), fminf(f[i].z, f[i].w)));
    vmax = fmaxf(vmax, fmaxf(fmaxf(f[i].x, f[i].y), fmaxf(f[i].z, f[i].w)));
  }
  vmin = wave_min_f(vmin);
  vmax = wave_max_f(vmax);
  if (lane == 0) { smin[wid] = vmin; smax[wid] = vmax; }
  __syncthreads();
#pragma unroll
  for (int i = 0; i < 8; ++i) {
    vmin = fminf(vmin, smin[i]);
    vmax = fmaxf(vmax, smax[i]);
  }

  const float d  = vmax - vmin;
  const float s  = (d > 0.f) ? d * (1.0f / 255.0f) : 1.0f;
  const float rs = 1.0f / s;
  const float zx = -128.0f - rintf(vmin * rs);

  int out[4];
#pragma unroll
  for (int i = 0; i < 4; ++i) {
    const float e[4] = {f[i].x, f[i].y, f[i].z, f[i].w};
    int b[4];
#pragma unroll
    for (int j = 0; j < 4; ++j) {
      float q = fminf(fmaxf(rintf(e[j] * rs) + zx, -128.0f), 127.0f);
      b[j] = (int)q;
    }
    out[i] = (b[0] & 0xFF) | ((b[1] & 0xFF) << 8) | ((b[2] & 0xFF) << 16) | ((b[3] & 0xFF) << 24);
  }
  *(int4*)(qx + (size_t)m * K_DIM + t * 16) = make_int4(out[0], out[1], out[2], out[3]);

  if (t == 0) {
    xs[m] = s;
    xo[m] = -s * zx;
  }
}

static __device__ __forceinline__ v8i load_a_frag(const int8_t* p) {
  const uint2 d0 = *(const uint2*)(p);
  const uint2 d1 = *(const uint2*)(p + 8);
  const uint2 d2 = *(const uint2*)(p + 32);
  const uint2 d3 = *(const uint2*)(p + 40);
  v8i a;
  a[0] = (int)d0.x; a[1] = (int)d0.y;
  a[2] = (int)d1.x; a[3] = (int)d1.y;
  a[4] = (int)d2.x; a[5] = (int)d2.y;
  a[6] = (int)d3.x; a[7] = (int)d3.y;
  return a;
}
static __device__ __forceinline__ v8i load_b_frag_lds(const int8_t* p) {
  const int4 d0 = *(const int4*)(p);
  const int4 d1 = *(const int4*)(p + 32);
  v8i b;
  b[0] = d0.x; b[1] = d0.y; b[2] = d0.z; b[3] = d0.w;
  b[4] = d1.x; b[5] = d1.y; b[6] = d1.z; b[7] = d1.w;
  return b;
}

__global__ __launch_bounds__(256, 1)
void gemm_iu8_kernel(const int8_t* __restrict__ qx,
                     const int8_t* __restrict__ qw,
                     const float* __restrict__ wscale,
                     const float* __restrict__ cw,
                     const float* __restrict__ xs,
                     const float* __restrict__ xo,
                     float* __restrict__ out) {
  __shared__ int8_t bsh[2][BROWS * BPITCH];

  const int t      = threadIdx.x;
  const int lane   = t & 31;
  const int wid    = t >> 5;
  const int lane16 = lane & 15;
  const int laneHi = lane >> 4;

  const int m0 = blockIdx.y * 128 + (wid >> 1) * 32;
  const int n0 = blockIdx.x * 64  + (wid & 1)  * 32;

  const int crow = t >> 2;
  const int ckol = (t & 3) * 32;
  const int8_t* csrc =
      qw + (size_t)(blockIdx.x * 64 + crow) * K_DIM + ckol;
  const uint32_t clds0 = (uint32_t)(uintptr_t)(&bsh[0][crow * BPITCH + ckol]);
  const uint32_t clds1 = (uint32_t)(uintptr_t)(&bsh[1][crow * BPITCH + ckol]);

  float sxe[2][8], oxe[2][8];
#pragma unroll
  for (int tm = 0; tm < 2; ++tm)
#pragma unroll
    for (int e = 0; e < 8; ++e) {
      const int m = m0 + tm * 16 + laneHi * 8 + e;
      sxe[tm][e] = xs[m];
      oxe[tm][e] = xo[m];
    }
  float cwv[2];
#pragma unroll
  for (int tn = 0; tn < 2; ++tn) cwv[tn] = cw[n0 + tn * 16 + lane16];

  const int8_t* pA[2];
  const float*  pS[2];
#pragma unroll
  for (int tm = 0; tm < 2; ++tm)
    pA[tm] = qx + (size_t)(m0 + tm * 16 + lane16) * K_DIM + laneHi * 16;
#pragma unroll
  for (int tn = 0; tn < 2; ++tn)
    pS[tn] = wscale + (size_t)(n0 + tn * 16 + lane16) * NG;

  const int nloc = (wid & 1) * 32;
  const int8_t* pBl[2];
#pragma unroll
  for (int tn = 0; tn < 2; ++tn)
    pBl[tn] = &bsh[0][(nloc + tn * 16 + lane16) * BPITCH + laneHi * 16];

  v8f facc[2][2];
#pragma unroll
  for (int tm = 0; tm < 2; ++tm)
#pragma unroll
    for (int tn = 0; tn < 2; ++tn) facc[tm][tn] = (v8f)(0.0f);

  asm volatile(
      "global_load_async_to_lds_b128 %0, %1, off\n\t"
      "global_load_async_to_lds_b128 %0, %1, off offset:16"
      :: "v"(clds0), "v"(csrc) : "memory");

  for (int g = 0; g < NG; ++g) {
    const int kb = g * GS;

    asm volatile("s_wait_asynccnt 0x0" ::: "memory");
    __syncthreads();

    if (g + 1 < NG) {
      const int8_t*  nsrc = csrc + (size_t)(kb + GS);
      const uint32_t nlds = ((g + 1) & 1) ? clds1 : clds0;
      asm volatile(
          "global_load_async_to_lds_b128 %0, %1, off\n\t"
          "global_load_async_to_lds_b128 %0, %1, off offset:16"
          :: "v"(nlds), "v"(nsrc) : "memory");
      asm volatile("global_prefetch_b8 %0, off" :: "v"(pA[0] + kb + GS));
      asm volatile("global_prefetch_b8 %0, off" :: "v"(pA[1] + kb + GS));
    }

    const int bufOff = (g & 1) * (BROWS * BPITCH);

    v8i iacc[2][2];
#pragma unroll
    for (int tm = 0; tm < 2; ++tm)
#pragma unroll
      for (int tn = 0; tn < 2; ++tn) iacc[tm][tn] = (v8i)(0);

#pragma unroll
    for (int ks = 0; ks < 2; ++ks) {
      v8i a[2], b[2];
#pragma unroll
      for (int tm = 0; tm < 2; ++tm) a[tm] = load_a_frag(pA[tm] + kb + ks * 64);
#pragma unroll
      for (int tn = 0; tn < 2; ++tn)
        b[tn] = load_b_frag_lds(pBl[tn] + bufOff + ks * 64);
#pragma unroll
      for (int tm = 0; tm < 2; ++tm)
#pragma unroll
        for (int tn = 0; tn < 2; ++tn)
          iacc[tm][tn] = __builtin_amdgcn_wmma_i32_16x16x64_iu8(
               true, a[tm],  true, b[tn],
              iacc[tm][tn],  false,  false);
    }

    const float sw[2] = {pS[0][g], pS[1][g]};
#pragma unroll
    for (int tm = 0; tm < 2; ++tm)
#pragma unroll
      for (int tn = 0; tn < 2; ++tn)
#pragma unroll
        for (int e = 0; e < 8; ++e)
          facc[tm][tn][e] += sw[tn] * (float)iacc[tm][tn][e];
  }

#pragma unroll
  for (int tm = 0; tm < 2; ++tm)
#pragma unroll
    for (int tn = 0; tn < 2; ++tn)
#pragma unroll
      for (int e = 0; e < 8; ++e) {
        const int m = m0 + tm * 16 + laneHi * 8 + e;
        const int n = n0 + tn * 16 + lane16;
        out[(size_t)m * N_DIM + n] =
            sxe[tm][e] * facc[tm][tn][e] + oxe[tm][e] * cwv[tn];
      }
}

extern "C" void kernel_launch(void* const* d_in, const int* in_sizes, int n_in,
                              void* d_out, int out_size, void* d_ws, size_t ws_size,
                              hipStream_t stream) {
  const float* x = (const float*)d_in[0];
  const float* w = (const float*)d_in[1];
  float* out = (float*)d_out;

  uint8_t* ws = (uint8_t*)d_ws;
  size_t o = 0;
  int8_t* qw = (int8_t*)(ws + o);           o += (size_t)N_DIM * K_DIM;
  int8_t* qx = (int8_t*)(ws + o);           o += (size_t)M_DIM * K_DIM;
  float* wscale = (float*)(ws + o);         o += (size_t)N_DIM * NG * 4;
  float* sqr    = (float*)(ws + o);         o += (size_t)N_DIM * NG * 4;
  float* cw     = (float*)(ws + o);         o += (size_t)N_DIM * 4;
  float* xs     = (float*)(ws + o);         o += (size_t)M_DIM * 4;
  float* xo     = (float*)(ws + o);         o += (size_t)M_DIM * 4;
  (void)o; (void)ws_size; (void)in_sizes; (void)n_in; (void)out_size;

  quant_w_kernel<<<dim3((N_DIM * NG) / 8), 256, 0, stream>>>(w, qw, wscale, sqr);
  cw_kernel<<<dim3(N_DIM / 256), 256, 0, stream>>>(wscale, sqr, cw);
  quant_x_kernel<<<dim3(M_DIM), 256, 0, stream>>>(x, qx, xs, xo);
  gemm_iu8_kernel<<<dim3(N_DIM / 64, M_DIM / 128), 256, 0, stream>>>(
      qx, qw, wscale, cw, xs, xo, out);
}
